// SentenceDecoder_3925600109279
// MI455X (gfx1250) — matrix operand order repaired and checked by reading, not compiled, not run on hardware
//
#include <hip/hip_runtime.h>
#include <math.h>

// ---------------------------------------------------------------------------
// Types for CDNA5 WMMA (wave32)
// ---------------------------------------------------------------------------
typedef __attribute__((ext_vector_type(16))) __bf16 v16bf;
typedef __attribute__((ext_vector_type(8)))  __bf16 v8bf;
typedef __attribute__((ext_vector_type(8)))  float  v8f;

union ABHalf { v16bf v; v8bf h[2]; };

// A fragment: 16x32 bf16 tile of row-major X[.., ld], tile origin (mBase, k0).
// Lanes 0-15: M=lane, elems 0..15 = K k0..k0+15.
// Lanes 16-31: M=lane-16, elems 0..15 = K k0+16..k0+31.
__device__ __forceinline__ v16bf load_a_frag(const __bf16* __restrict__ X, int ld,
                                             int row, int k0, int half) {
  ABHalf u;
  const __bf16* p = X + (size_t)row * ld + k0 + half * 16;
  u.h[0] = *(const v8bf*)(p);
  u.h[1] = *(const v8bf*)(p + 8);
  return u.v;
}

// B fragment: 32x16 bf16 tile of W^T where W is row-major [N, K].
// Lane = column n; lanes 0-15 hold K k0..k0+15, lanes 16-31 hold K k0+16..k0+31.
__device__ __forceinline__ v16bf load_b_frag(const __bf16* __restrict__ W, int ld,
                                             int col, int k0, int half) {
  return *(const v16bf*)(W + (size_t)col * ld + k0 + half * 16);
}

#define NPW 4  // N-tiles per wave (A-fragment reuse)

// ---------------------------------------------------------------------------
// Gate GEMM: G[64, 16*NT] = X[64,Kx] @ Wx^T + Hh[64,Kh] @ Wh^T   (bf16 in, f32 out)
// One wave computes a 16(M) x 64(N) strip. mTile = wave&3, nGroup = wave>>2.
// ---------------------------------------------------------------------------
__global__ void __launch_bounds__(256)
gates_gemm(const __bf16* __restrict__ X, int Kx, const __bf16* __restrict__ Wx,
           const __bf16* __restrict__ Hh, int Kh, const __bf16* __restrict__ Wh,
           float* __restrict__ G, int NT)
{
  const int lane  = threadIdx.x & 31;
  const int gw    = (blockIdx.x * blockDim.x + threadIdx.x) >> 5;
  const int mTile = gw & 3;
  const int nGrp  = gw >> 2;
  if (nGrp >= NT / NPW) return;           // wave-uniform: EXEC stays all-ones
  const int half  = lane >> 4;
  const int lrow  = lane & 15;
  const int rowA  = mTile * 16 + lrow;
  const int nT0   = nGrp * NPW;

  v8f acc[NPW];
  #pragma unroll
  for (int j = 0; j < NPW; ++j) acc[j] = v8f{};

  for (int k0 = 0; k0 < Kx; k0 += 32) {
    v16bf a = load_a_frag(X, Kx, rowA, k0, half);
    #pragma unroll
    for (int j = 0; j < NPW; ++j) {
      v16bf b = load_b_frag(Wx, Kx, (nT0 + j) * 16 + lrow, k0, half);
      acc[j] = __builtin_amdgcn_wmma_f32_16x16x32_bf16(false, a, false, b,
                                                       (short)0, acc[j], false, false);
    }
  }
  for (int k0 = 0; k0 < Kh; k0 += 32) {
    v16bf a = load_a_frag(Hh, Kh, rowA, k0, half);
    #pragma unroll
    for (int j = 0; j < NPW; ++j) {
      v16bf b = load_b_frag(Wh, Kh, (nT0 + j) * 16 + lrow, k0, half);
      acc[j] = __builtin_amdgcn_wmma_f32_16x16x32_bf16(false, a, false, b,
                                                       (short)0, acc[j], false, false);
    }
  }

  const int ldg   = NT * 16;
  const int rbase = mTile * 16 + half * 8;   // C layout: VGPR r -> row rbase+r
  #pragma unroll
  for (int j = 0; j < NPW; ++j) {
    const int col = (nT0 + j) * 16 + lrow;
    #pragma unroll
    for (int r = 0; r < 8; ++r)
      G[(size_t)(rbase + r) * ldg + col] = acc[j][r];
  }
}

// ---------------------------------------------------------------------------
// Logits GEMM + fused per-tile row argmax.
// tv/ti: [64, NT] per-tile (max value, global argmax index).
// ---------------------------------------------------------------------------
__global__ void __launch_bounds__(256)
logits_argmax_tiles(const __bf16* __restrict__ Hin, int K,
                    const __bf16* __restrict__ Wo, const float* __restrict__ bout,
                    float* __restrict__ tv, int* __restrict__ ti, int NT)
{
  const int lane  = threadIdx.x & 31;
  const int gw    = (blockIdx.x * blockDim.x + threadIdx.x) >> 5;
  const int mTile = gw & 3;
  const int nGrp  = gw >> 2;
  if (nGrp >= NT / NPW) return;
  const int half  = lane >> 4;
  const int lrow  = lane & 15;
  const int rowA  = mTile * 16 + lrow;
  const int nT0   = nGrp * NPW;

  v8f acc[NPW];
  #pragma unroll
  for (int j = 0; j < NPW; ++j) acc[j] = v8f{};

  for (int k0 = 0; k0 < K; k0 += 32) {
    v16bf a = load_a_frag(Hin, K, rowA, k0, half);
    #pragma unroll
    for (int j = 0; j < NPW; ++j) {
      v16bf b = load_b_frag(Wo, K, (nT0 + j) * 16 + lrow, k0, half);
      acc[j] = __builtin_amdgcn_wmma_f32_16x16x32_bf16(false, a, false, b,
                                                       (short)0, acc[j], false, false);
    }
  }

  #pragma unroll
  for (int j = 0; j < NPW; ++j) {
    const int myCol = (nT0 + j) * 16 + lrow;
    const float bo  = bout[myCol];
    #pragma unroll
    for (int r = 0; r < 8; ++r) {
      float v  = acc[j][r] + bo;
      int   bi = myCol;
      // reduce across the 16 lanes of each half-wave (xor masks < 16 keep halves apart)
      #pragma unroll
      for (int msk = 1; msk < 16; msk <<= 1) {
        float ov = __shfl_xor(v, msk, 32);
        int   oi = __shfl_xor(bi, msk, 32);
        if (ov > v || (ov == v && oi < bi)) { v = ov; bi = oi; }
      }
      if (lrow == 0) {
        const int row = mTile * 16 + half * 8 + r;
        tv[(size_t)row * NT + (nT0 + j)] = v;
        ti[(size_t)row * NT + (nT0 + j)] = bi;
      }
    }
  }
}

// ---------------------------------------------------------------------------
// LSTM pointwise: gate order i,f,g,o (torch). G is [64, 4*1024].
// ---------------------------------------------------------------------------
__global__ void __launch_bounds__(256)
lstm_pointwise(const float* __restrict__ G, const float* __restrict__ bih,
               const float* __restrict__ bhh, float* __restrict__ c,
               __bf16* __restrict__ hb)
{
  const int Hd  = 1024;
  const int idx = blockIdx.x * blockDim.x + threadIdx.x;   // 64*1024 threads
  const int n   = idx & (Hd - 1);
  const int m   = idx >> 10;
  const float* g = G + (size_t)m * 4 * Hd;
  const float gi = g[n]          + bih[n]          + bhh[n];
  const float gf = g[Hd + n]     + bih[Hd + n]     + bhh[Hd + n];
  const float gc = g[2 * Hd + n] + bih[2 * Hd + n] + bhh[2 * Hd + n];
  const float go = g[3 * Hd + n] + bih[3 * Hd + n] + bhh[3 * Hd + n];
  const float si = 1.f / (1.f + __expf(-gi));
  const float sf = 1.f / (1.f + __expf(-gf));
  const float so = 1.f / (1.f + __expf(-go));
  const float cn = sf * c[idx] + si * tanhf(gc);
  const float hn = so * tanhf(cn);
  c[idx]  = cn;
  hb[idx] = (__bf16)hn;
}

// ---------------------------------------------------------------------------
// Final argmax over per-tile candidates, emit token id, gather next embedding.
// One block per batch row.
// ---------------------------------------------------------------------------
__global__ void __launch_bounds__(256)
argmax_gather(const float* __restrict__ tv, const int* __restrict__ ti, int NT,
              const float* __restrict__ embed, __bf16* __restrict__ xb,
              int* __restrict__ outIds, int t, int T)
{
  __shared__ float sv[256];
  __shared__ int   si[256];
  const int row = blockIdx.x;
  const int tid = threadIdx.x;
  float best = -INFINITY;
  int   bidx = 0x7fffffff;
  for (int j = tid; j < NT; j += 256) {
    const float v = tv[(size_t)row * NT + j];
    const int   i = ti[(size_t)row * NT + j];
    if (v > best || (v == best && i < bidx)) { best = v; bidx = i; }
  }
  sv[tid] = best; si[tid] = bidx;
  __syncthreads();
  for (int s = 128; s > 0; s >>= 1) {
    if (tid < s) {
      if (sv[tid + s] > sv[tid] || (sv[tid + s] == sv[tid] && si[tid + s] < si[tid])) {
        sv[tid] = sv[tid + s]; si[tid] = si[tid + s];
      }
    }
    __syncthreads();
  }
  const int pred = si[0];
  if (tid == 0) outIds[row * T + t] = pred;
  for (int d = tid; d < 512; d += 256)
    xb[(size_t)row * 512 + d] = (__bf16)embed[(size_t)pred * 512 + d];
}

// ---------------------------------------------------------------------------
// Helpers: fp32 -> bf16 weight conversion; state init.
// ---------------------------------------------------------------------------
__global__ void __launch_bounds__(256)
cvt_bf16(const float* __restrict__ s, __bf16* __restrict__ d, int n)
{
  const int i = blockIdx.x * blockDim.x + threadIdx.x;
  if (i < n) d[i] = (__bf16)s[i];
}

__global__ void __launch_bounds__(256)
init_state(const float* __restrict__ features, __bf16* __restrict__ xb,
           float* __restrict__ c0, float* __restrict__ c1,
           __bf16* __restrict__ h0b, __bf16* __restrict__ h1b)
{
  const int idx = blockIdx.x * blockDim.x + threadIdx.x;   // 64*1024 threads
  c0[idx] = 0.f; c1[idx] = 0.f;
  h0b[idx] = (__bf16)0.f; h1b[idx] = (__bf16)0.f;
  if (idx < 64 * 512) xb[idx] = (__bf16)features[idx];
}

// ---------------------------------------------------------------------------
// Host launcher
// ---------------------------------------------------------------------------
extern "C" void kernel_launch(void* const* d_in, const int* in_sizes, int n_in,
                              void* d_out, int out_size, void* d_ws, size_t ws_size,
                              hipStream_t stream)
{
  const int B = 64, D = 512, H = 1024, V = 32000, T = 32;
  const int NT_G = (4 * H) / 16;   // 256 N-tiles for gate GEMMs
  const int NT_V = V / 16;         // 2000 N-tiles for logits

  const float* features = (const float*)d_in[0];
  const float* embed    = (const float*)d_in[1];
  const float* Wih0     = (const float*)d_in[2];
  const float* Whh0     = (const float*)d_in[3];
  const float* bih0     = (const float*)d_in[4];
  const float* bhh0     = (const float*)d_in[5];
  const float* Wih1     = (const float*)d_in[6];
  const float* Whh1     = (const float*)d_in[7];
  const float* bih1     = (const float*)d_in[8];
  const float* bhh1     = (const float*)d_in[9];
  const float* Wout     = (const float*)d_in[10];
  const float* bout     = (const float*)d_in[11];

  char*  base = (char*)d_ws;
  size_t off  = 0;
  auto carve = [&](size_t bytes) -> char* {
    off = (off + 255) & ~(size_t)255;
    char* p = base + off;
    off += bytes;
    return p;
  };

  __bf16* wih0b = (__bf16*)carve((size_t)4 * H * D * 2);
  __bf16* whh0b = (__bf16*)carve((size_t)4 * H * H * 2);
  __bf16* wih1b = (__bf16*)carve((size_t)4 * H * H * 2);
  __bf16* whh1b = (__bf16*)carve((size_t)4 * H * H * 2);
  __bf16* woutb = (__bf16*)carve((size_t)V * H * 2);
  __bf16* xb    = (__bf16*)carve((size_t)B * D * 2);
  __bf16* h0b   = (__bf16*)carve((size_t)B * H * 2);
  __bf16* h1b   = (__bf16*)carve((size_t)B * H * 2);
  float*  c0    = (float*)carve((size_t)B * H * 4);
  float*  c1    = (float*)carve((size_t)B * H * 4);
  float*  g     = (float*)carve((size_t)B * 4 * H * 4);
  float*  tv    = (float*)carve((size_t)B * NT_V * 4);
  int*    ti    = (int*)carve((size_t)B * NT_V * 4);
  (void)ws_size; (void)in_sizes; (void)n_in; (void)out_size;

  int* outIds = (int*)d_out;

  // One-time (per launch) bf16 weight conversion: makes all weights ~95 MB,
  // L2-resident across the 32 sequential steps.
  auto cvt = [&](const float* s, __bf16* d, int n) {
    cvt_bf16<<<(n + 255) / 256, 256, 0, stream>>>(s, d, n);
  };
  cvt(Wih0, wih0b, 4 * H * D);
  cvt(Whh0, whh0b, 4 * H * H);
  cvt(Wih1, wih1b, 4 * H * H);
  cvt(Whh1, whh1b, 4 * H * H);
  cvt(Wout, woutb, V * H);

  init_state<<<(B * H) / 256, 256, 0, stream>>>(features, xb, c0, c1, h0b, h1b);

  const int gatesBlocks  = (4 * (NT_G / NPW) + 7) / 8;   // 8 waves/block
  const int logitsBlocks = (4 * (NT_V / NPW) + 7) / 8;

  for (int t = 0; t < T; ++t) {
    // Layer 0: gates = x @ Wih0^T + h0 @ Whh0^T
    gates_gemm<<<gatesBlocks, 256, 0, stream>>>(xb, D, wih0b, h0b, H, whh0b, g, NT_G);
    lstm_pointwise<<<(B * H) / 256, 256, 0, stream>>>(g, bih0, bhh0, c0, h0b);
    // Layer 1: gates = h0 @ Wih1^T + h1 @ Whh1^T
    gates_gemm<<<gatesBlocks, 256, 0, stream>>>(h0b, H, wih1b, h1b, H, whh1b, g, NT_G);
    lstm_pointwise<<<(B * H) / 256, 256, 0, stream>>>(g, bih1, bhh1, c1, h1b);
    // Logits + fused tile argmax
    logits_argmax_tiles<<<logitsBlocks, 256, 0, stream>>>(h1b, H, woutb, bout, tv, ti, NT_V);
    // Final argmax, emit token, gather next embedding
    argmax_gather<<<B, 256, 0, stream>>>(tv, ti, NT_V, embed, xb, outIds, t, T);
  }
}
